// ContinuousEpisodicVLM_34041910788243
// MI455X (gfx1250) — matrix operand order repaired and checked by reading, not compiled, not run on hardware
//
#include <hip/hip_runtime.h>
#include <math.h>

// ---------------- problem constants (from reference) ----------------
#define DIM      768
#define P_PATCH  2048
#define M_NODES  65536
#define C_CLS    1000
#define C_PAD    1024      // anchors padded to multiple of 16 tiles
#define KTOP     50
#define TAU      0.02f
#define TAU_CONF 0.8f

typedef __bf16 bf16;
typedef __attribute__((ext_vector_type(16))) bf16  v16bf;
typedef __attribute__((ext_vector_type(8)))  bf16  v8bf;
typedef __attribute__((ext_vector_type(8)))  float v8f;
typedef __attribute__((ext_vector_type(4)))  int   v4i;

// ---------------- async-to-LDS helpers (CDNA5) ----------------
// Copy 16 bytes global -> LDS via the async engine (tracked with ASYNCcnt).
// Builtin prototype (from hipcc diagnostic): (v4i __global*, v4i __local*, Ii, Ii)
__device__ inline void async_copy_b128(const bf16* gsrc, bf16* ldst) {
#if __has_builtin(__builtin_amdgcn_global_load_async_to_lds_b128)
    __builtin_amdgcn_global_load_async_to_lds_b128(
        (__attribute__((address_space(1))) v4i*)gsrc,
        (__attribute__((address_space(3))) v4i*)ldst, 0, 0);
#else
    // flat pointer low 32 bits == LDS byte offset for shared objects
    unsigned loff = (unsigned)(uintptr_t)ldst;
    asm volatile("global_load_async_to_lds_b128 %0, %1, off"
                 :: "v"(loff), "v"(gsrc) : "memory");
#endif
}
template <int N>
__device__ inline void async_wait() {
#if __has_builtin(__builtin_amdgcn_s_wait_asynccnt)
    __builtin_amdgcn_s_wait_asynccnt((unsigned short)N);
#else
    if (N == 0) asm volatile("s_wait_asynccnt 0x0" ::: "memory");
    else        asm volatile("s_wait_asynccnt 0x8" ::: "memory");
#endif
}

// ---------------- block reductions ----------------
template <int NT>
__device__ inline float blk_sum(float v, float* red) {
    int tid = threadIdx.x;
    red[tid] = v; __syncthreads();
    for (int s = NT / 2; s > 0; s >>= 1) {
        if (tid < s) red[tid] += red[tid + s];
        __syncthreads();
    }
    float r = red[0]; __syncthreads();
    return r;
}
template <int NT>
__device__ inline float blk_max(float v, float* red) {
    int tid = threadIdx.x;
    red[tid] = v; __syncthreads();
    for (int s = NT / 2; s > 0; s >>= 1) {
        if (tid < s) red[tid] = fmaxf(red[tid], red[tid + s]);
        __syncthreads();
    }
    float r = red[0]; __syncthreads();
    return r;
}

// ---------------- WMMA fragment helpers ----------------
// A fragment (16-bit A 16x32 layout): lane holds row (lane&15); K-chunk
// {kbase + 16*(lane>>4)}, 16 contiguous bf16 (matches B fragment order).
__device__ inline v16bf load_a_frag_lds(const bf16* As, int kbase, int lane) {
    int row  = lane & 15;
    int aoff = kbase + ((lane >> 4) << 4);
    v8bf a0 = *(const v8bf*)(As + row * DIM + aoff);
    v8bf a1 = *(const v8bf*)(As + row * DIM + aoff + 8);
    return __builtin_shufflevector(a0, a1, 0,1,2,3,4,5,6,7,8,9,10,11,12,13,14,15);
}
// B fragment (16-bit B 32x16): lane holds column (lane&15) = one memory row;
// K-chunk {kbase + 16*(lane>>4)}, 16 contiguous bf16. Global-memory version.
__device__ inline v16bf load_b_frag_glb(const bf16* B, int colBase, int kbase, int lane) {
    size_t row = (size_t)(colBase + (lane & 15));
    return *(const v16bf*)(B + row * DIM + kbase + ((lane >> 4) << 4));
}
// Same fragment out of an LDS-staged [128 x 128] bf16 sub-tile.
__device__ inline v16bf load_b_frag_lds(const bf16* Bs, int colLocal, int kLocal, int lane) {
    int row = colLocal + (lane & 15);
    return *(const v16bf*)(Bs + row * 128 + kLocal + ((lane >> 4) << 4));
}

// ---------------- one-time conversion kernels ----------------
__global__ void k_cvt_f2b(const float* __restrict__ src, bf16* __restrict__ dst, int n) {
    for (int i = blockIdx.x * blockDim.x + threadIdx.x; i < n; i += gridDim.x * blockDim.x)
        dst[i] = (bf16)src[i];
}
__global__ void k_cvt_anchors(const float* __restrict__ src, bf16* __restrict__ dst) {
    int n = C_PAD * DIM;
    for (int i = blockIdx.x * blockDim.x + threadIdx.x; i < n; i += gridDim.x * blockDim.x) {
        int c = i / DIM;
        dst[i] = (c < C_CLS) ? (bf16)src[i] : (bf16)0.0f;
    }
}
__global__ void k_init_patches(const float* __restrict__ src,
                               float* __restrict__ pf, bf16* __restrict__ pb, int n) {
    for (int i = blockIdx.x * blockDim.x + threadIdx.x; i < n; i += gridDim.x * blockDim.x) {
        float v = src[i];
        pf[i] = v;
        pb[i] = (bf16)v;
    }
}

// ---------------- prototypes + System-1 logits ----------------
__global__ __launch_bounds__(256)
void k_protos(const float* __restrict__ tg, const float* __restrict__ anch,
              const float* __restrict__ csum, const float* __restrict__ ccnt,
              float* __restrict__ protos, float* __restrict__ d_out) {
    __shared__ float red[256];
    int c = blockIdx.x, tid = threadIdx.x;
    float cnt = fmaxf(ccnt[c], 1.0f);
    float v[3], ss = 0.f;
#pragma unroll
    for (int j = 0; j < 3; j++) {
        int d = tid + j * 256;
        float x = csum[(size_t)c * DIM + d] / cnt;
        v[j] = x; ss += x * x;
    }
    ss = blk_sum<256>(ss, red);
    float rn = 1.0f / fmaxf(sqrtf(ss), 1e-12f);
    float t[3]; ss = 0.f;
#pragma unroll
    for (int j = 0; j < 3; j++) {
        int d = tid + j * 256;
        float x = anch[(size_t)c * DIM + d] + v[j] * rn;   // ALPHA = 1.0
        t[j] = x; ss += x * x;
    }
    ss = blk_sum<256>(ss, red);
    float rn2 = 1.0f / fmaxf(sqrtf(ss), 1e-12f);
    float dt = 0.f;
#pragma unroll
    for (int j = 0; j < 3; j++) {
        int d = tid + j * 256;
        float pr = t[j] * rn2;
        protos[(size_t)c * DIM + d] = pr;
        dt += pr * tg[d];
    }
    dt = blk_sum<256>(dt, red);
    if (tid == 0) d_out[c] = 100.0f * dt;
}

// entropy of sys1 logits -> active flag, step = 0
__global__ __launch_bounds__(1024)
void k_init_state(const float* __restrict__ d_out_logits, float* __restrict__ d_out,
                  int* __restrict__ active, int* __restrict__ step) {
    __shared__ float red[1024];
    int tid = threadIdx.x;
    float lv = (tid < C_CLS) ? d_out_logits[tid] : -1e30f;
    float m  = blk_max<1024>(lv, red);
    float e  = (tid < C_CLS) ? expf(lv - m) : 0.f;
    float es = blk_sum<1024>(e, red);
    float pc = e / es;
    float ent = blk_sum<1024>((tid < C_CLS) ? -pc * logf(pc + 1e-10f) : 0.f, red);
    if (tid == 0) {
        *active = (ent > TAU_CONF) ? 1 : 0;
        *step = 0;
        d_out[C_CLS] = 0.0f;   // step output slot
    }
}

// ---------------- fused WMMA GEMM + top-50 retrieval ----------------
// grid: P/16 blocks, 256 threads (8 waves). Per 128-column chunk the K=768
// reduction is split into 6 sub-tiles of 128; the B sub-tile (128x128 bf16,
// 32 KB) is double-buffered in LDS and filled by async-to-LDS b128 copies
// (8 per thread per tile -> s_wait_asynccnt <= 8 drains exactly the previous
// tile). WMMAs then read both A and B from LDS.
#define NKT 6                       // K sub-tiles per chunk (768 / 128)
#define NTILES (512 * NKT)          // total sub-tiles over M

__device__ inline void issue_tile(const bf16* __restrict__ memB, bf16* Bs, int i, int tid) {
    int chunk = i / NKT, t = i - chunk * NKT;
    int col0 = chunk * 128, kk = t * 128;
#pragma unroll
    for (int j = 0; j < 8; ++j) {
        int seg = tid + j * 256;        // 2048 b128 segments per tile
        int row = seg >> 4;             // 16 segments per 128-elem row
        int s   = seg & 15;
        async_copy_b128(memB + (size_t)(col0 + row) * DIM + kk + s * 8,
                        Bs + row * 128 + s * 8);
    }
}

__global__ __launch_bounds__(256)
void k_topk(const bf16* __restrict__ patchesB, const bf16* __restrict__ memB,
            float* __restrict__ topv, int* __restrict__ topi,
            const int* __restrict__ active) {
    if (*active == 0) return;
    __shared__ __align__(32) bf16  As[16 * DIM];       // 24 KB patch tile
    __shared__ __align__(32) bf16  Bs[2][128 * 128];   // 64 KB double buffer
    __shared__ float Cs[8 * 256];                      //  8 KB staged sim tiles
    __shared__ float tv[16 * KTOP];
    __shared__ int   ti[16 * KTOP];
    int tid = threadIdx.x, lane = tid & 31, wave = tid >> 5;
    int rowBase = blockIdx.x * 16;

    for (int i = tid; i < 16 * DIM; i += 256)
        As[i] = patchesB[(size_t)(rowBase + i / DIM) * DIM + (i % DIM)];
    for (int i = tid; i < 16 * KTOP; i += 256) { tv[i] = -1e30f; ti[i] = 0; }

    issue_tile(memB, Bs[0], 0, tid);        // prime the pipeline
    __syncthreads();

    float minv = -1e30f; int minp = 0;      // registers of owner threads

    for (int chunk = 0; chunk < 512; ++chunk) {
        v8f acc = {};
        for (int t = 0; t < NKT; ++t) {
            int i = chunk * NKT + t;
            if (i + 1 < NTILES) { issue_tile(memB, Bs[(i + 1) & 1], i + 1, tid); async_wait<8>(); }
            else                { async_wait<0>(); }
            __syncthreads();
            const bf16* Bb = Bs[i & 1];
#pragma unroll
            for (int k8 = 0; k8 < 4; ++k8) {
                v16bf a = load_a_frag_lds(As, t * 128 + k8 * 32, lane);
                v16bf b = load_b_frag_lds(Bb, wave * 16, k8 * 32, lane);
                acc = __builtin_amdgcn_wmma_f32_16x16x32_bf16(
                    false, a, false, b, (short)0, acc, false, false);
            }
            __syncthreads();   // protect Bs[(i+2)&1] refill vs. readers
        }
        {   // stage C tile:  M = r + 8*(lane>>4), N = lane&15
            int n = lane & 15, mh = lane >> 4;
#pragma unroll
            for (int r = 0; r < 8; r++)
                Cs[wave * 256 + (r + 8 * mh) * 16 + n] = acc[r];
        }
        __syncthreads();
        if (tid < 16) {                    // one owner thread per patch row
            int row = tid, col0 = chunk * 128;
            for (int j = 0; j < 128; j++) {
                float v = Cs[(j >> 4) * 256 + row * 16 + (j & 15)];
                if (v > minv) {
                    tv[row * KTOP + minp] = v;
                    ti[row * KTOP + minp] = col0 + j;
                    minv = tv[row * KTOP]; minp = 0;
                    for (int q = 1; q < KTOP; q++) {
                        float x = tv[row * KTOP + q];
                        if (x < minv) { minv = x; minp = q; }
                    }
                }
            }
        }
        __syncthreads();
    }
    if (tid < 16)
        for (int q = 0; q < KTOP; q++) {
            topv[(size_t)(rowBase + tid) * KTOP + q] = tv[tid * KTOP + q];
            topi[(size_t)(rowBase + tid) * KTOP + q] = ti[tid * KTOP + q];
        }
}

// ---------------- softmax-weighted gather + renormalize ----------------
__global__ __launch_bounds__(128)
void k_message(const float* __restrict__ pcur, const float* __restrict__ topv,
               const int* __restrict__ topi, const float* __restrict__ memF,
               float* __restrict__ pnewF, bf16* __restrict__ pnewB,
               const int* __restrict__ active) {
    if (*active == 0) return;
    __shared__ float w[KTOP];
    __shared__ int   id[KTOP];
    __shared__ float red[128];
    int p = blockIdx.x, tid = threadIdx.x;
    if (tid < KTOP) {
        int ix = topi[(size_t)p * KTOP + tid];
        id[tid] = ix;
        __builtin_prefetch(memF + (size_t)ix * DIM, 0, 1);  // global_prefetch_b8
    }
    if (tid == 0) {
        float mx = -1e30f;
        for (int q = 0; q < KTOP; q++) mx = fmaxf(mx, topv[(size_t)p * KTOP + q]);
        float s = 0.f;
        for (int q = 0; q < KTOP; q++) {
            float e = expf((topv[(size_t)p * KTOP + q] - mx) / TAU);
            w[q] = e; s += e;
        }
        float inv = 1.0f / s;
        for (int q = 0; q < KTOP; q++) w[q] *= inv;
    }
    __syncthreads();
    float acc[6], ss = 0.f;
#pragma unroll
    for (int j = 0; j < 6; j++) {
        int d = tid + j * 128;
        float a = pcur[(size_t)p * DIM + d];
        for (int q = 0; q < KTOP; q++)
            a += w[q] * memF[(size_t)id[q] * DIM + d];
        acc[j] = a; ss += a * a;
    }
    ss = blk_sum<128>(ss, red);
    float rn = 1.0f / fmaxf(sqrtf(ss), 1e-12f);
#pragma unroll
    for (int j = 0; j < 6; j++) {
        int d = tid + j * 128;
        float v = acc[j] * rn;
        pnewF[(size_t)p * DIM + d] = v;
        pnewB[(size_t)p * DIM + d] = (bf16)v;
    }
}

// gated commit: patches = where(active, new_patches, patches)
__global__ void k_commit(const float* __restrict__ pnewF, const bf16* __restrict__ pnewB,
                         float* __restrict__ pcurF, bf16* __restrict__ pcurB,
                         const int* __restrict__ active, int n) {
    if (*active == 0) return;
    for (int i = blockIdx.x * blockDim.x + threadIdx.x; i < n; i += gridDim.x * blockDim.x) {
        pcurF[i] = pnewF[i];
        pcurB[i] = pnewB[i];
    }
}

// ---------------- WMMA evidence: ev[p] = max_c new_patch . anchor_c ----------------
__global__ __launch_bounds__(256)
void k_evidence(const bf16* __restrict__ pnewB, const bf16* __restrict__ anchB,
                float* __restrict__ ev, const int* __restrict__ active) {
    if (*active == 0) return;
    __shared__ bf16  As[16 * DIM];
    __shared__ float Cs[8 * 256];
    int tid = threadIdx.x, lane = tid & 31, wave = tid >> 5;
    int rowBase = blockIdx.x * 16;
    for (int i = tid; i < 16 * DIM; i += 256)
        As[i] = pnewB[(size_t)(rowBase + i / DIM) * DIM + (i % DIM)];
    __syncthreads();
    float rm = -1e30f;                     // owner threads (tid < 16)
    for (int col0 = 0; col0 < C_PAD; col0 += 128) {
        int colW = col0 + wave * 16;
        v8f acc = {};
#pragma unroll 4
        for (int k = 0; k < DIM; k += 32) {
            v16bf a = load_a_frag_lds(As, k, lane);
            v16bf b = load_b_frag_glb(anchB, colW, k, lane);
            acc = __builtin_amdgcn_wmma_f32_16x16x32_bf16(
                false, a, false, b, (short)0, acc, false, false);
        }
        {
            int n = lane & 15, mh = lane >> 4;
#pragma unroll
            for (int r = 0; r < 8; r++)
                Cs[wave * 256 + (r + 8 * mh) * 16 + n] = acc[r];
        }
        __syncthreads();
        if (tid < 16)
            for (int j = 0; j < 128; j++)
                if (col0 + j < C_CLS)
                    rm = fmaxf(rm, Cs[(j >> 4) * 256 + tid * 16 + (j & 15)]);
        __syncthreads();
    }
    if (tid < 16) ev[rowBase + tid] = rm;
}

// ---------------- pooling, logits, entropy, state update ----------------
__global__ __launch_bounds__(1024)
void k_pool(const float* __restrict__ pnewF, const float* __restrict__ ev,
            const float* __restrict__ protos, float* __restrict__ d_out,
            int* __restrict__ active, int* __restrict__ step) {
    if (*active == 0) return;
    __shared__ float red[1024];
    __shared__ float g[DIM];
    __shared__ float lg[C_CLS];
    int tid = threadIdx.x;
    float m = -1e30f;
    for (int i = tid; i < P_PATCH; i += 1024) m = fmaxf(m, ev[i]);
    m = blk_max<1024>(m, red);
    float s = 0.f;
    for (int i = tid; i < P_PATCH; i += 1024) s += expf((ev[i] - m) / TAU);
    s = blk_sum<1024>(s, red);
    if (tid < DIM) {
        float acc = 0.f;
        for (int p = 0; p < P_PATCH; p++)
            acc += expf((ev[p] - m) / TAU) * pnewF[(size_t)p * DIM + tid];
        g[tid] = acc / s;
    }
    __syncthreads();
    float ss = (tid < DIM) ? g[tid] * g[tid] : 0.f;
    ss = blk_sum<1024>(ss, red);
    float rn = 1.0f / fmaxf(sqrtf(ss), 1e-12f);
    if (tid < DIM) g[tid] *= rn;
    __syncthreads();
    if (tid < C_CLS) {
        float acc = 0.f;
        for (int d = 0; d < DIM; d++) acc += g[d] * protos[(size_t)tid * DIM + d];
        lg[tid] = 100.0f * acc;
    }
    __syncthreads();
    float lv = (tid < C_CLS) ? lg[tid] : -1e30f;
    float lm = blk_max<1024>(lv, red);
    float e  = (tid < C_CLS) ? expf(lv - lm) : 0.f;
    float es = blk_sum<1024>(e, red);
    float pc = e / es;
    float ent = blk_sum<1024>((tid < C_CLS) ? -pc * logf(pc + 1e-10f) : 0.f, red);
    if (tid < C_CLS) d_out[tid] = lg[tid];
    if (tid == 0) {
        int ns = *step + 1;
        *step = ns;
        d_out[C_CLS] = (float)ns;
        *active = (ent > TAU_CONF) ? 1 : 0;
    }
}

// ---------------- host side ----------------
static inline size_t align_up(size_t x, size_t a) { return (x + a - 1) & ~(a - 1); }

extern "C" void kernel_launch(void* const* d_in, const int* in_sizes, int n_in,
                              void* d_out, int out_size, void* d_ws, size_t ws_size,
                              hipStream_t stream) {
    (void)in_sizes; (void)n_in; (void)out_size; (void)ws_size;
    const float* tg    = (const float*)d_in[0];   // [1,768]
    const float* tp    = (const float*)d_in[1];   // [2048,768]
    const float* memF  = (const float*)d_in[2];   // [65536,768]
    const float* anch  = (const float*)d_in[3];   // [1000,768]
    const float* csum  = (const float*)d_in[4];   // [1000,768]
    const float* ccnt  = (const float*)d_in[5];   // [1000]
    float* out = (float*)d_out;                   // [1000 logits][1 step]

    char*  base = (char*)d_ws;
    size_t off  = 0;
    auto carve = [&](size_t bytes) { void* p = base + off; off = align_up(off + bytes, 256); return p; };
    int*   active  = (int*)  carve(sizeof(int));
    int*   step    = (int*)  carve(sizeof(int));
    bf16*  memB    = (bf16*) carve((size_t)M_NODES * DIM * sizeof(bf16));   // 100.7 MB (< L2)
    bf16*  anchB   = (bf16*) carve((size_t)C_PAD   * DIM * sizeof(bf16));
    float* protos  = (float*)carve((size_t)C_CLS   * DIM * sizeof(float));
    float* pcurF   = (float*)carve((size_t)P_PATCH * DIM * sizeof(float));
    bf16*  pcurB   = (bf16*) carve((size_t)P_PATCH * DIM * sizeof(bf16));
    float* pnewF   = (float*)carve((size_t)P_PATCH * DIM * sizeof(float));
    bf16*  pnewB   = (bf16*) carve((size_t)P_PATCH * DIM * sizeof(bf16));
    float* topv    = (float*)carve((size_t)P_PATCH * KTOP * sizeof(float));
    int*   topi    = (int*)  carve((size_t)P_PATCH * KTOP * sizeof(int));
    float* ev      = (float*)carve((size_t)P_PATCH * sizeof(float));

    k_cvt_f2b     <<<8192, 256, 0, stream>>>(memF, memB, M_NODES * DIM);
    k_cvt_anchors <<<3072, 256, 0, stream>>>(anch, anchB);
    k_init_patches<<<6144, 256, 0, stream>>>(tp, pcurF, pcurB, P_PATCH * DIM);
    k_protos      <<<C_CLS, 256, 0, stream>>>(tg, anch, csum, ccnt, protos, out);
    k_init_state  <<<1, 1024, 0, stream>>>(out, out, active, step);

    for (int it = 0; it < 3; ++it) {      // max_steps = 3 in the reference setup
        k_topk    <<<P_PATCH / 16, 256, 0, stream>>>(pcurB, memB, topv, topi, active);
        k_message <<<P_PATCH, 128, 0, stream>>>(pcurF, topv, topi, memF, pnewF, pnewB, active);
        k_commit  <<<6144, 256, 0, stream>>>(pnewF, pnewB, pcurF, pcurB, active, P_PATCH * DIM);
        k_evidence<<<P_PATCH / 16, 256, 0, stream>>>(pnewB, anchB, ev, active);
        k_pool    <<<1, 1024, 0, stream>>>(pnewF, ev, protos, out, active, step);
    }
}
